// IterRefineODEMS_48421461295722
// MI455X (gfx1250) — matrix operand order repaired and checked by reading, not compiled, not run on hardware
//
#include <hip/hip_runtime.h>
#include <math.h>

// ---------------------------------------------------------------------------
// CDNA5 (gfx1250) implementation of the two-level 3D flow refinement network.
// Convolutions are implicit-GEMM on V_WMMA_F32_16X16X32_BF16 (bf16 in, f32 acc)
// with a (tap, cin)-ordered K dimension (no divides in the inner loop) and all
// Cout tiles held in one wave (MT accumulators) so each B fragment feeds MT
// WMMAs instead of one.
// ---------------------------------------------------------------------------

typedef __attribute__((ext_vector_type(16))) __bf16 v16bf;
typedef __attribute__((ext_vector_type(8)))  __bf16 v8bf;
typedef __attribute__((ext_vector_type(8)))  float  v8f;

// ------------------------- elementwise helpers -----------------------------

__global__ void k_zero(float* p, long n) {
  long i = blockIdx.x * (long)blockDim.x + threadIdx.x;
  long st = (long)gridDim.x * blockDim.x;
  for (; i < n; i += st) p[i] = 0.f;
}

__global__ void k_copy(const float* __restrict__ s, float* __restrict__ d, long n) {
  long i = blockIdx.x * (long)blockDim.x + threadIdx.x;
  long st = (long)gridDim.x * blockDim.x;
  for (; i < n; i += st) d[i] = s[i];
}

// ---------------------- trilinear resize (factor 0.5) ----------------------
__global__ void k_resize_half(const float* __restrict__ in, float* __restrict__ out,
                              int C, int Di, int Hi, int Wi) {
  int Do = Di / 2, Ho = Hi / 2, Wo = Wi / 2;
  long S = (long)Do * Ho * Wo;
  long i = blockIdx.x * (long)blockDim.x + threadIdx.x;
  if (i >= (long)C * S) return;
  int c = (int)(i / S);
  long v = i % S;
  int x = (int)(v % Wo), y = (int)((v / Wo) % Ho), z = (int)(v / ((long)Wo * Ho));
  const float* ch = in + (long)c * Di * Hi * Wi;
  int HWi = Hi * Wi;
  float s = 0.f;
#pragma unroll
  for (int dz = 0; dz < 2; ++dz)
#pragma unroll
    for (int dy = 0; dy < 2; ++dy)
#pragma unroll
      for (int dx = 0; dx < 2; ++dx)
        s += ch[(2 * z + dz) * HWi + (2 * y + dy) * Wi + (2 * x + dx)];
  out[i] = s * 0.0625f; // (1/8) * 0.5
}

// ---------------------- trilinear resize (factor 2) ------------------------
__global__ void k_resize_double(const float* __restrict__ in, float* __restrict__ out,
                                int C, int Di, int Hi, int Wi, float scale) {
  int Do = 2 * Di, Ho = 2 * Hi, Wo = 2 * Wi;
  long S = (long)Do * Ho * Wo;
  long i = blockIdx.x * (long)blockDim.x + threadIdx.x;
  if (i >= (long)C * S) return;
  int c = (int)(i / S);
  long v = i % S;
  int x = (int)(v % Wo), y = (int)((v / Wo) % Ho), z = (int)(v / ((long)Wo * Ho));
  float sz = z * 0.5f - 0.25f, sy = y * 0.5f - 0.25f, sx = x * 0.5f - 0.25f;
  float z0f = floorf(sz), y0f = floorf(sy), x0f = floorf(sx);
  float fz = sz - z0f, fy = sy - y0f, fx = sx - x0f;
  int z0 = (int)z0f, y0 = (int)y0f, x0 = (int)x0f;
  int z1 = z0 + 1, y1 = y0 + 1, x1 = x0 + 1;
  z0 = min(max(z0, 0), Di - 1); z1 = min(max(z1, 0), Di - 1);
  y0 = min(max(y0, 0), Hi - 1); y1 = min(max(y1, 0), Hi - 1);
  x0 = min(max(x0, 0), Wi - 1); x1 = min(max(x1, 0), Wi - 1);
  const float* ch = in + (long)c * Di * Hi * Wi;
  int HWi = Hi * Wi;
  float c00 = ch[z0 * HWi + y0 * Wi + x0] * (1 - fx) + ch[z0 * HWi + y0 * Wi + x1] * fx;
  float c01 = ch[z0 * HWi + y1 * Wi + x0] * (1 - fx) + ch[z0 * HWi + y1 * Wi + x1] * fx;
  float c10 = ch[z1 * HWi + y0 * Wi + x0] * (1 - fx) + ch[z1 * HWi + y0 * Wi + x1] * fx;
  float c11 = ch[z1 * HWi + y1 * Wi + x0] * (1 - fx) + ch[z1 * HWi + y1 * Wi + x1] * fx;
  float c0 = c00 * (1 - fy) + c01 * fy;
  float c1 = c10 * (1 - fy) + c11 * fy;
  out[i] = (c0 * (1 - fz) + c1 * fz) * scale;
}

// ------------------------------- warping -----------------------------------
__global__ void k_warp(const float* __restrict__ vol, const float* __restrict__ flow,
                       float* __restrict__ out, int C, int D, int H, int W, int add_flow) {
  long S = (long)D * H * W;
  long v = blockIdx.x * (long)blockDim.x + threadIdx.x;
  if (v >= S) return;
  int x = (int)(v % W), y = (int)((v / W) % H), z = (int)(v / ((long)W * H));
  float cz = z + flow[v], cy = y + flow[S + v], cx = x + flow[2 * S + v];
  float z0f = floorf(cz), y0f = floorf(cy), x0f = floorf(cx);
  float fz = cz - z0f, fy = cy - y0f, fx = cx - x0f;
  int z0 = (int)z0f, y0 = (int)y0f, x0 = (int)x0f;
  float wgt[8];
  long idx[8];
  int HW = H * W, t = 0;
#pragma unroll
  for (int dz = 0; dz < 2; ++dz)
#pragma unroll
    for (int dy = 0; dy < 2; ++dy)
#pragma unroll
      for (int dx = 0; dx < 2; ++dx) {
        int zi = z0 + dz, yi = y0 + dy, xi = x0 + dx;
        float w = (dz ? fz : 1.f - fz) * (dy ? fy : 1.f - fy) * (dx ? fx : 1.f - fx);
        bool ok = (zi >= 0) & (zi < D) & (yi >= 0) & (yi < H) & (xi >= 0) & (xi < W);
        wgt[t] = ok ? w : 0.f;
        int zc = min(max(zi, 0), D - 1), yc = min(max(yi, 0), H - 1), xc = min(max(xi, 0), W - 1);
        idx[t] = (long)zc * HW + yc * W + xc;
        ++t;
      }
  for (int c = 0; c < C; ++c) {
    const float* ch = vol + (long)c * S;
    float acc = 0.f;
#pragma unroll
    for (int k = 0; k < 8; ++k) acc += wgt[k] * ch[idx[k]];
    if (add_flow) acc += flow[(long)c * S + v];
    out[(long)c * S + v] = acc;
  }
}

// -------------------- deterministic two-stage reduction --------------------
__global__ void k_reduce_partial(const float* __restrict__ a, long n, float* __restrict__ part) {
  __shared__ float sh[512];
  float s = 0.f, q = 0.f;
  for (long i = blockIdx.x * (long)blockDim.x + threadIdx.x; i < n;
       i += (long)gridDim.x * blockDim.x) {
    float v = a[i];
    s += v; q += v * v;
  }
  int tid = threadIdx.x;
  sh[tid] = s; sh[256 + tid] = q;
  __syncthreads();
  for (int o = 128; o > 0; o >>= 1) {
    if (tid < o) { sh[tid] += sh[tid + o]; sh[256 + tid] += sh[256 + tid + o]; }
    __syncthreads();
  }
  if (tid == 0) { part[2 * blockIdx.x] = sh[0]; part[2 * blockIdx.x + 1] = sh[256]; }
}

__global__ void k_stats_final(const float* __restrict__ pA, const float* __restrict__ pB,
                              int nb, float nA, float nB, float* __restrict__ st) {
  __shared__ float sh[1024];
  int tid = threadIdx.x;
  float sA = 0.f, qA = 0.f, sB = 0.f, qB = 0.f;
  for (int i = tid; i < nb; i += 256) {
    sA += pA[2 * i]; qA += pA[2 * i + 1];
    sB += pB[2 * i]; qB += pB[2 * i + 1];
  }
  sh[tid] = sA; sh[256 + tid] = qA; sh[512 + tid] = sB; sh[768 + tid] = qB;
  __syncthreads();
  for (int o = 128; o > 0; o >>= 1) {
    if (tid < o) {
      sh[tid] += sh[tid + o]; sh[256 + tid] += sh[256 + tid + o];
      sh[512 + tid] += sh[512 + tid + o]; sh[768 + tid] += sh[768 + tid + o];
    }
    __syncthreads();
  }
  if (tid == 0) {
    float mA = sh[0] / nA, mB = sh[512] / nB;
    float vA = (sh[256] - nA * mA * mA) / (nA - 1.f);
    float vB = (sh[768] - nB * mB * mB) / (nB - 1.f);
    float sdev = 0.5f * (sqrtf(fmaxf(vA, 0.f)) + sqrtf(fmaxf(vB, 0.f)));
    st[0] = 0.5f * (mA + mB);
    st[1] = 1.f / sdev;
  }
}

// ------------------------------ cost volume --------------------------------
__global__ void k_cost_volume(const float* __restrict__ f1, const float* __restrict__ f2,
                              const float* __restrict__ st, float* __restrict__ out,
                              int C, int md, int D, int H, int W) {
  int n = 2 * md + 1, n3 = n * n * n;
  long S = (long)D * H * W;
  long i = blockIdx.x * (long)blockDim.x + threadIdx.x;
  if (i >= (long)n3 * S) return;
  int off = (int)(i / S);
  long v = i % S;
  int x = (int)(v % W), y = (int)((v / W) % H), z = (int)(v / ((long)W * H));
  int dz = off / (n * n) - md, dy = (off / n) % n - md, dx = off % n - md;
  int zz = z + dz, yy = y + dy, xx = x + dx;
  float acc = 0.f;
  if ((unsigned)zz < (unsigned)D && (unsigned)yy < (unsigned)H && (unsigned)xx < (unsigned)W) {
    float m = st[0], is = st[1];
    long v2 = (long)zz * H * W + yy * W + xx;
    for (int c = 0; c < C; ++c)
      acc += (f1[(long)c * S + v] - m) * (f2[(long)c * S + v2] - m);
    acc *= is * is / (float)C;
  }
  out[i] = acc >= 0.f ? acc : 0.05f * acc; // leaky 0.05
}

// -------------------- weight convert: f32 -> padded bf16 -------------------
// src: [Cout][Cin][27]  (flat layout of [Cout,Cin,3,3,3])
// dst: [CoutPad][27][Cpad] bf16, zero padded (rows >= Cout, cin >= Cin).
__global__ void k_wcvt(const float* __restrict__ w, __bf16* __restrict__ o,
                       int Cout, int Cin, int CoutPad, int Cpad) {
  long total = (long)CoutPad * 27 * Cpad;
  long i = blockIdx.x * (long)blockDim.x + threadIdx.x;
  if (i >= total) return;
  int r = (int)(i / (27 * Cpad));
  int rem = (int)(i % (27 * Cpad));
  int tap = rem / Cpad;
  int c = rem % Cpad;
  float v = (r < Cout && c < Cin) ? w[((long)r * Cin + c) * 27 + tap] : 0.f;
  o[i] = (__bf16)v;
}

// ---------------- implicit-GEMM conv3d (3x3x3, SAME) via WMMA --------------
// One wave: (MT*16 Cout) x 16(voxel) tile with MT f32 accumulator fragments.
// Loop over 27 taps; per tap, loop cin in 32-blocks. Each B fragment (one
// 32-cin block of 16 voxels) is loaded once and feeds MT back-to-back
// v_wmma_f32_16x16x32_bf16 against MT A fragments (2 x 16B loads each).
// W is a multiple of 16, so a tile stays in one row: (z,y) are wave-uniform
// -> the (zz,yy) bounds test is a uniform branch (EXEC all-0 or all-1).
template <int D, int H, int W, int MT>
__launch_bounds__(128)
__global__ void k_conv3d_wmma(const float* __restrict__ in, const __bf16* __restrict__ wt,
                              const float* __restrict__ bias, float* __restrict__ out,
                              int Cin, int Cpad, int Cout, int leaky) {
  constexpr int S = D * H * W;
  constexpr int HW = H * W;
  const int lane = threadIdx.x & 31;
  const int wave = threadIdx.x >> 5;
  const int n_tile = blockIdx.x * 4 + wave;
  const int half = lane >> 4;
  const int ln = lane & 15;
  const int n = n_tile * 16 + ln;
  const int x = n % W;
  const int y = (n / W) % H;
  const int z = n / HW;
  const int Kp = 27 * Cpad;
  // A row for m_tile 0; other m_tiles offset by mt*16*Kp.
  const __bf16* wrow = wt + (long)ln * Kp + half * 16;
  v8f acc[MT] = {};
  for (int kd = 0; kd < 3; ++kd) {
    const int zz = z + kd - 1;
#pragma unroll
    for (int kh = 0; kh < 3; ++kh) {
      const int yy = y + kh - 1;
      const bool zyok = ((unsigned)zz < (unsigned)D) & ((unsigned)yy < (unsigned)H);
#pragma unroll
      for (int kw = 0; kw < 3; ++kw) {
        if (zyok) { // wave-uniform
          const int tap = kd * 9 + kh * 3 + kw;
          const int xx = x + kw - 1;
          const bool xok = (unsigned)xx < (unsigned)W;
          const int xc = min(max(xx, 0), W - 1);
          const float* bp0 = in + (long)zz * HW + (long)yy * W + xc + (long)(half * 16) * S;
          const __bf16* wp = wrow + (long)tap * Cpad;
          for (int c0 = 0; c0 < Cpad; c0 += 32) {
            // ---- B fragment: cin = c0 + half*16 + e, stride-S loads ----
            v16bf b;
            if (c0 + 32 <= Cin) { // uniform fast path: no channel clamp
              const float* bp = bp0 + (long)c0 * S;
#pragma unroll
              for (int e = 0; e < 16; ++e) {
                float v = bp[(long)e * S];
                b[e] = (__bf16)(xok ? v : 0.f);
              }
            } else {              // tail block: padded weights are zero
              const int cb = c0 + half * 16;
#pragma unroll
              for (int e = 0; e < 16; ++e) {
                int cc = min(cb + e, Cin - 1);
                float v = in[(long)cc * S + (long)zz * HW + (long)yy * W + xc];
                b[e] = (__bf16)(xok ? v : 0.f);
              }
            }
            // ---- MT A fragments, MT WMMAs sharing this B ----
#pragma unroll
            for (int mt = 0; mt < MT; ++mt) {
              const __bf16* wq = wp + (long)mt * 16 * Kp + c0;
              const v8bf alo = *(const v8bf*)(wq);
              const v8bf ahi = *(const v8bf*)(wq + 8);
              v16bf a;
#pragma unroll
              for (int e = 0; e < 8; ++e) { a[e] = alo[e]; a[e + 8] = ahi[e]; }
              acc[mt] = __builtin_amdgcn_wmma_f32_16x16x32_bf16(false, a, false, b,
                                                                (short)0, acc[mt],
                                                                false, false);
            }
          }
        }
      }
    }
  }
  const int mb = half * 8;
#pragma unroll
  for (int mt = 0; mt < MT; ++mt) {
#pragma unroll
    for (int r = 0; r < 8; ++r) {
      int m = mt * 16 + mb + r;
      if (m < Cout) {
        float vv = acc[mt][r] + bias[m];
        if (leaky) vv = vv >= 0.f ? vv : 0.02f * vv;
        out[(long)m * S + n] = vv;
      }
    }
  }
}

// ------------------------------ final vf -----------------------------------
__global__ void k_final_vf(const float* __restrict__ d2, const float* __restrict__ vu,
                           const float* __restrict__ du, const float* __restrict__ tptr,
                           float* __restrict__ out, int D, int H, int W) {
  long S = (long)D * H * W;
  long v = blockIdx.x * (long)blockDim.x + threadIdx.x;
  if (v >= S) return;
  int x = (int)(v % W), y = (int)((v / W) % H), z = (int)(v / ((long)W * H));
  float cz = z + vu[v], cy = y + vu[S + v], cx = x + vu[2 * S + v];
  float z0f = floorf(cz), y0f = floorf(cy), x0f = floorf(cx);
  float fz = cz - z0f, fy = cy - y0f, fx = cx - x0f;
  int z0 = (int)z0f, y0 = (int)y0f, x0 = (int)x0f;
  float wgt[8];
  long idx[8];
  int HW = H * W, t = 0;
#pragma unroll
  for (int dz = 0; dz < 2; ++dz)
#pragma unroll
    for (int dy = 0; dy < 2; ++dy)
#pragma unroll
      for (int dx = 0; dx < 2; ++dx) {
        int zi = z0 + dz, yi = y0 + dy, xi = x0 + dx;
        float w = (dz ? fz : 1.f - fz) * (dy ? fy : 1.f - fy) * (dx ? fx : 1.f - fx);
        bool ok = (zi >= 0) & (zi < D) & (yi >= 0) & (yi < H) & (xi >= 0) & (xi < W);
        wgt[t] = ok ? w : 0.f;
        int zc = min(max(zi, 0), D - 1), yc = min(max(yi, 0), H - 1), xc = min(max(xi, 0), W - 1);
        idx[t] = (long)zc * HW + yc * W + xc;
        ++t;
      }
  float inv = 1.f / (1.f - tptr[0]);
  for (int c = 0; c < 3; ++c) {
    const float* ch = d2 + (long)c * S;
    float acc = 0.f;
#pragma unroll
    for (int k = 0; k < 8; ++k) acc += wgt[k] * ch[idx[k]];
    out[(long)c * S + v] = (acc + vu[(long)c * S + v] - du[(long)c * S + v]) * inv;
  }
}

// ---------------------------------------------------------------------------
// Host side
// ---------------------------------------------------------------------------

struct LayerDef { int wi, bi, Cin, Cpad, Cout, CoutPad; };

extern "C" void kernel_launch(void* const* d_in, const int* in_sizes, int n_in,
                              void* d_out, int out_size, void* d_ws, size_t ws_size,
                              hipStream_t stream) {
  (void)in_sizes; (void)n_in; (void)ws_size;
  const int D1 = 48, H1 = 40, W1 = 48;
  const int D2 = 96, H2 = 80, W2 = 96;
  const long S1 = (long)D1 * H1 * W1;   // 92160
  const long S2 = (long)D2 * H2 * W2;   // 737280

  const float* t_in     = (const float*)d_in[0];
  const float* disp_up  = (const float*)d_in[1];
  const float* feat_x   = (const float*)d_in[2];
  const float* feat_y   = (const float*)d_in[3];
  const float* feat_xu  = (const float*)d_in[4];
  const float* feat_yu  = (const float*)d_in[5];
  const float* ctx      = (const float*)d_in[6];
  float* out = (float*)d_out;

  // ---- workspace carve-up (256B aligned) ----
  size_t off = 0;
  auto carve = [&](size_t bytes) -> char* {
    off = (off + 255) & ~(size_t)255;
    char* p = (char*)d_ws + off;
    off += bytes;
    return p;
  };
  float* x1    = (float*)carve(sizeof(float) * 384 * S1); // level-1 concat + context
  float* x2    = (float*)carve(sizeof(float) * 126 * S2); // level-2 concat + hidden
  float* vel1  = (float*)carve(sizeof(float) * 3 * S1);
  float* velup = (float*)carve(sizeof(float) * 3 * S2);
  float* vup2  = (float*)carve(sizeof(float) * 3 * S2);
  float* st1   = (float*)carve(sizeof(float) * 8);
  float* st2   = (float*)carve(sizeof(float) * 8);
  float* partA = (float*)carve(sizeof(float) * 2048);
  float* partB = (float*)carve(sizeof(float) * 2048);

  static const LayerDef L1[5] = {
    {7,  8, 224, 224, 64, 64}, {9, 10, 288, 288, 48, 48},
    {11,12, 336, 352, 32, 32}, {13,14, 368, 384, 16, 16},
    {15,16, 16, 32, 3, 16}};
  static const LayerDef L2[3] = {
    {17,18, 78, 96, 32, 32}, {19,20, 110, 128, 16, 16},
    {21,22, 16, 32, 3, 16}};
  __bf16* wbf[8];
  {
    int j = 0;
    for (int i = 0; i < 5; ++i) wbf[j++] = (__bf16*)carve(2UL * L1[i].CoutPad * 27 * L1[i].Cpad);
    for (int i = 0; i < 3; ++i) wbf[j++] = (__bf16*)carve(2UL * L2[i].CoutPad * 27 * L2[i].Cpad);
  }

  const int TB = 256, RB = 1024;
  auto blks = [&](long n) { return (int)((n + TB - 1) / TB); };

  auto run_conv1 = [&](const LayerDef& L, const __bf16* w, const float* in_p,
                       const float* b_p, float* out_p, int leaky) {
    dim3 grid((unsigned)(S1 / 64));
    switch (L.CoutPad / 16) {
      case 4: k_conv3d_wmma<48, 40, 48, 4><<<grid, 128, 0, stream>>>(in_p, w, b_p, out_p, L.Cin, L.Cpad, L.Cout, leaky); break;
      case 3: k_conv3d_wmma<48, 40, 48, 3><<<grid, 128, 0, stream>>>(in_p, w, b_p, out_p, L.Cin, L.Cpad, L.Cout, leaky); break;
      case 2: k_conv3d_wmma<48, 40, 48, 2><<<grid, 128, 0, stream>>>(in_p, w, b_p, out_p, L.Cin, L.Cpad, L.Cout, leaky); break;
      default: k_conv3d_wmma<48, 40, 48, 1><<<grid, 128, 0, stream>>>(in_p, w, b_p, out_p, L.Cin, L.Cpad, L.Cout, leaky); break;
    }
  };
  auto run_conv2 = [&](const LayerDef& L, const __bf16* w, const float* in_p,
                       const float* b_p, float* out_p, int leaky) {
    dim3 grid((unsigned)(S2 / 64));
    switch (L.CoutPad / 16) {
      case 2: k_conv3d_wmma<96, 80, 96, 2><<<grid, 128, 0, stream>>>(in_p, w, b_p, out_p, L.Cin, L.Cpad, L.Cout, leaky); break;
      default: k_conv3d_wmma<96, 80, 96, 1><<<grid, 128, 0, stream>>>(in_p, w, b_p, out_p, L.Cin, L.Cpad, L.Cout, leaky); break;
    }
  };

  // ---- weight conversion (bf16, [CoutPad][27][Cpad], zero-padded) ----
  for (int i = 0; i < 5; ++i) {
    long tot = (long)L1[i].CoutPad * 27 * L1[i].Cpad;
    k_wcvt<<<blks(tot), TB, 0, stream>>>((const float*)d_in[L1[i].wi], wbf[i],
                                         L1[i].Cout, L1[i].Cin, L1[i].CoutPad, L1[i].Cpad);
  }
  for (int i = 0; i < 3; ++i) {
    long tot = (long)L2[i].CoutPad * 27 * L2[i].Cpad;
    k_wcvt<<<blks(tot), TB, 0, stream>>>((const float*)d_in[L2[i].wi], wbf[5 + i],
                                         L2[i].Cout, L2[i].Cin, L2[i].CoutPad, L2[i].Cpad);
  }

  // ---- zero the (mostly-zero) output tuple ----
  k_zero<<<4096, TB, 0, stream>>>(out, (long)out_size);

  // ================= level 1 (48x40x48) =================
  k_resize_half<<<blks(3 * S1), TB, 0, stream>>>(disp_up, x1 + 221 * S1, 3, D2, H2, W2);
  k_warp<<<blks(S1), TB, 0, stream>>>(feat_x, x1 + 221 * S1, x1 + 157 * S1, 32, D1, H1, W1, 0);
  k_reduce_partial<<<RB, TB, 0, stream>>>(x1 + 157 * S1, 32 * S1, partA);
  k_reduce_partial<<<RB, TB, 0, stream>>>(feat_y, 32 * S1, partB);
  k_stats_final<<<1, TB, 0, stream>>>(partA, partB, RB, (float)(32 * S1), (float)(32 * S1), st1);
  k_cost_volume<<<blks(125 * S1), TB, 0, stream>>>(feat_y, x1 + 157 * S1, st1, x1,
                                                   32, 2, D1, H1, W1);
  k_copy<<<blks(32 * S1), TB, 0, stream>>>(feat_y, x1 + 125 * S1, 32 * S1);
  k_copy<<<blks(32 * S1), TB, 0, stream>>>(ctx, x1 + 189 * S1, 32 * S1);
  run_conv1(L1[0], wbf[0], x1, (const float*)d_in[L1[0].bi], x1 + 224 * S1, 1);
  run_conv1(L1[1], wbf[1], x1, (const float*)d_in[L1[1].bi], x1 + 288 * S1, 1);
  run_conv1(L1[2], wbf[2], x1, (const float*)d_in[L1[2].bi], x1 + 336 * S1, 1);
  run_conv1(L1[3], wbf[3], x1, (const float*)d_in[L1[3].bi], x1 + 368 * S1, 1);
  run_conv1(L1[4], wbf[4], x1 + 368 * S1, (const float*)d_in[L1[4].bi], vel1, 0);

  // ================= upsample to level 2 (96x80x96) =================
  k_resize_double<<<blks(16 * S2), TB, 0, stream>>>(x1 + 368 * S1, x2 + 59 * S2, 16, D1, H1, W1, 1.f);
  k_resize_double<<<blks(3 * S2), TB, 0, stream>>>(vel1, velup, 3, D1, H1, W1, 2.f);
  k_warp<<<blks(S2), TB, 0, stream>>>(disp_up, velup, x2 + 75 * S2, 3, D2, H2, W2, 1);
  k_warp<<<blks(S2), TB, 0, stream>>>(feat_xu, x2 + 75 * S2, x2 + 43 * S2, 16, D2, H2, W2, 0);
  k_copy<<<blks(16 * S2), TB, 0, stream>>>(feat_yu, x2 + 27 * S2, 16 * S2);
  k_reduce_partial<<<RB, TB, 0, stream>>>(x2 + 43 * S2, 16 * S2, partA);
  k_reduce_partial<<<RB, TB, 0, stream>>>(feat_yu, 16 * S2, partB);
  k_stats_final<<<1, TB, 0, stream>>>(partA, partB, RB, (float)(16 * S2), (float)(16 * S2), st2);
  k_cost_volume<<<blks(27 * S2), TB, 0, stream>>>(feat_yu, x2 + 43 * S2, st2, x2,
                                                  16, 1, D2, H2, W2);
  run_conv2(L2[0], wbf[5], x2, (const float*)d_in[L2[0].bi], x2 + 78 * S2, 1);
  run_conv2(L2[1], wbf[6], x2, (const float*)d_in[L2[1].bi], x2 + 110 * S2, 1);
  run_conv2(L2[2], wbf[7], x2 + 110 * S2, (const float*)d_in[L2[2].bi], vup2, 0);

  // vf = (warp(disp_up2, vup2) + vup2 - disp_up) / (1 - t)
  k_final_vf<<<blks(S2), TB, 0, stream>>>(x2 + 75 * S2, vup2, disp_up, t_in, out, D2, H2, W2);
}
